// LinearAttention_34230889349620
// MI455X (gfx1250) — compile-verified, matrix operand order checked by reading, not run on hardware
//
#include <hip/hip_runtime.h>
#include <hip/hip_bf16.h>

typedef __attribute__((ext_vector_type(16))) __bf16 v16bf;
typedef __attribute__((ext_vector_type(8)))  float  v8f;

#define D_IN 512
#define D_K  64
#define ROWS_P1 256   // rows per phase-1 block (8 waves * 32 rows)

union Frag16 {
  v16bf v;
  __bf16         h[16];
  unsigned short u[16];
  unsigned int   w[8];
  uint4          q[2];
};

__device__ __forceinline__ __bf16 tobf(float x) { return (__bf16)x; }

__device__ __forceinline__ float phi_elu1(float x) {
  return x > 0.0f ? x + 1.0f : __expf(x);   // elu(x)+1
}

// ---------------------------------------------------------------------------
// Kernel 1: convert W_q/W_k/W_v (fp32 [512][64]) into bf16 B-fragments laid
// out per-lane for v_wmma_f32_16x16x32_bf16: Bfrag[frag(12)][ks(16)][lane(32)][16]
// B element (pair v, half o2) of lane L:
//   kk = ks*32 + (v>=4)*16 + (L>=16)*8 + (v%4)*2 + o2 ,  col = ntile*16 + (L%16)
// ---------------------------------------------------------------------------
__global__ void prep_w_kernel(const float* __restrict__ Wq,
                              const float* __restrict__ Wk,
                              const float* __restrict__ Wv,
                              __bf16* __restrict__ Bfrag) {
  int o = blockIdx.x * blockDim.x + threadIdx.x;   // 98304 total
  int i    = o & 15;
  int lane = (o >> 4) & 31;
  int ks   = (o >> 9) & 15;
  int frag = o >> 13;
  int v  = i >> 1, o2 = i & 1, g = v >> 2, vv = v & 3;
  int half = lane >> 4, nn = lane & 15;
  int kk  = ks * 32 + g * 16 + half * 8 + vv * 2 + o2;
  int col = (frag & 3) * 16 + nn;
  int m3  = frag >> 2;
  const float* W = (m3 == 0) ? Wq : (m3 == 1) ? Wk : Wv;
  Bfrag[o] = tobf(W[kk * D_K + col]);
}

// ---------------------------------------------------------------------------
// Kernel 2: fused projections. Each wave computes TWO 16-row tiles (32 rows)
// so every B-fragment load (2 x b128) feeds two WMMAs. A fragments load as
// 4 x b128 of contiguous fp32 + native bf16 converts. K_phi/V staged in two
// 128-row LDS passes (64 KB), rank-128 VALU updates -> per-block partials.
// ---------------------------------------------------------------------------
__global__ void __launch_bounds__(256) phase1_kernel(
    const float* __restrict__ X,
    const __bf16* __restrict__ Bfrag,
    __bf16* __restrict__ Qphi,
    float* __restrict__ partial) {
  __shared__ float Ks[128][64];
  __shared__ float Vs[128][64];

  const int tid  = threadIdx.x;
  const int lane = tid & 31;
  const int wv   = tid >> 5;
  const int m    = lane & 15;
  const int half = lane >> 4;

  const int rowBase = blockIdx.x * ROWS_P1 + wv * 32;
  v8f acc0[12] = {};
  v8f acc1[12] = {};
  const float* xr0 = X + (size_t)(rowBase + m) * D_IN;
  const float* xr1 = xr0 + 16 * D_IN;

  for (int ks = 0; ks < 16; ++ks) {
    Frag16 A0, A1;
#pragma unroll
    for (int g = 0; g < 2; ++g) {
      const int base = ks * 32 + g * 16 + half * 8;
      float4 a = *reinterpret_cast<const float4*>(xr0 + base);
      float4 b = *reinterpret_cast<const float4*>(xr0 + base + 4);
      float4 c = *reinterpret_cast<const float4*>(xr1 + base);
      float4 d = *reinterpret_cast<const float4*>(xr1 + base + 4);
      A0.h[8*g+0] = tobf(a.x); A0.h[8*g+1] = tobf(a.y);
      A0.h[8*g+2] = tobf(a.z); A0.h[8*g+3] = tobf(a.w);
      A0.h[8*g+4] = tobf(b.x); A0.h[8*g+5] = tobf(b.y);
      A0.h[8*g+6] = tobf(b.z); A0.h[8*g+7] = tobf(b.w);
      A1.h[8*g+0] = tobf(c.x); A1.h[8*g+1] = tobf(c.y);
      A1.h[8*g+2] = tobf(c.z); A1.h[8*g+3] = tobf(c.w);
      A1.h[8*g+4] = tobf(d.x); A1.h[8*g+5] = tobf(d.y);
      A1.h[8*g+6] = tobf(d.z); A1.h[8*g+7] = tobf(d.w);
    }
#pragma unroll
    for (int f = 0; f < 12; ++f) {
      Frag16 B;
      const uint4* bp = reinterpret_cast<const uint4*>(
          Bfrag + (size_t)((f * 16 + ks) * 32 + lane) * 16);
      B.q[0] = bp[0];
      B.q[1] = bp[1];
      acc0[f] = __builtin_amdgcn_wmma_f32_16x16x32_bf16(
          false, A0.v, false, B.v, (short)0, acc0[f], false, false);
      acc1[f] = __builtin_amdgcn_wmma_f32_16x16x32_bf16(
          false, A1.v, false, B.v, (short)0, acc1[f], false, false);
    }
  }

  // phi on Q (acc 0..3) and K (acc 4..7) for both tiles
#pragma unroll
  for (int f = 0; f < 8; ++f)
#pragma unroll
    for (int j = 0; j < 8; ++j) {
      acc0[f][j] = phi_elu1(acc0[f][j]);
      acc1[f][j] = phi_elu1(acc1[f][j]);
    }

  // Q_phi -> global bf16 row-major. D layout: lane col n=m, VGPR r -> row r+8*half.
#pragma unroll
  for (int t = 0; t < 4; ++t)
#pragma unroll
    for (int r = 0; r < 8; ++r) {
      int row0 = rowBase + r + 8 * half;
      Qphi[(size_t)row0 * D_K + t * 16 + m]        = tobf(acc0[t][r]);
      Qphi[(size_t)(row0 + 16) * D_K + t * 16 + m] = tobf(acc1[t][r]);
    }

  // KV partial: two staging passes of 128 rows through 64 KB LDS
  const int d1 = tid >> 2;          // KV row owned by this thread (0..63)
  const int c0 = (tid & 3) << 4;    // KV col base (0,16,32,48)
  float kv[16];
  float ksum = 0.0f;
#pragma unroll
  for (int i = 0; i < 16; ++i) kv[i] = 0.0f;

  for (int pass = 0; pass < 2; ++pass) {
    if ((wv >> 2) == pass) {
      const int rl = (wv & 3) * 32 + 8 * half;
#pragma unroll
      for (int t = 0; t < 4; ++t)
#pragma unroll
        for (int r = 0; r < 8; ++r) {
          Ks[rl + r][t * 16 + m]      = acc0[4 + t][r];
          Vs[rl + r][t * 16 + m]      = acc0[8 + t][r];
          Ks[rl + 16 + r][t * 16 + m] = acc1[4 + t][r];
          Vs[rl + 16 + r][t * 16 + m] = acc1[8 + t][r];
        }
    }
    __syncthreads();
    for (int r = 0; r < 128; ++r) {
      float kp = Ks[r][d1];
      const float4* vp = reinterpret_cast<const float4*>(&Vs[r][c0]);
      float4 a = vp[0], b = vp[1], c = vp[2], d = vp[3];
      kv[0]  += kp * a.x;  kv[1]  += kp * a.y;  kv[2]  += kp * a.z;  kv[3]  += kp * a.w;
      kv[4]  += kp * b.x;  kv[5]  += kp * b.y;  kv[6]  += kp * b.z;  kv[7]  += kp * b.w;
      kv[8]  += kp * c.x;  kv[9]  += kp * c.y;  kv[10] += kp * c.z;  kv[11] += kp * c.w;
      kv[12] += kp * d.x;  kv[13] += kp * d.y;  kv[14] += kp * d.z;  kv[15] += kp * d.w;
      ksum   += kp;
    }
    __syncthreads();
  }

  // write per-block partial: [64*64 KV floats][64 ksum floats] = 4160 floats
  float* pb = partial + (size_t)blockIdx.x * 4160;
  float4* pv = reinterpret_cast<float4*>(pb + d1 * 64 + c0);
  pv[0] = make_float4(kv[0],  kv[1],  kv[2],  kv[3]);
  pv[1] = make_float4(kv[4],  kv[5],  kv[6],  kv[7]);
  pv[2] = make_float4(kv[8],  kv[9],  kv[10], kv[11]);
  pv[3] = make_float4(kv[12], kv[13], kv[14], kv[15]);
  if ((tid & 3) == 0) pb[4096 + d1] = ksum;
}

// ---------------------------------------------------------------------------
// Kernel 3: reduce block partials into KVext[64][80] (cols 0..63 = KV,
// col 64 = k_sum, cols 65..79 = 0), written directly as bf16 B-fragments:
// B2[f=nt*2+ks][lane][16] with nt in 0..4, ks in 0..1.
// ---------------------------------------------------------------------------
__global__ void reduce_kernel(const float* __restrict__ partial, int nblk,
                              __bf16* __restrict__ B2) {
  int o = blockIdx.x * blockDim.x + threadIdx.x;   // 5120 total
  if (o >= 5 * 2 * 32 * 16) return;
  int i    = o & 15;
  int lane = (o >> 4) & 31;
  int ks   = (o >> 9) & 1;
  int nt   = o >> 10;
  int v = i >> 1, o2 = i & 1, g = v >> 2, vv = v & 3;
  int half = lane >> 4, nn = lane & 15;
  int kk  = ks * 32 + g * 16 + half * 8 + vv * 2 + o2;
  int col = nt * 16 + nn;
  float s = 0.0f;
  if (col < 64) {
    for (int b = 0; b < nblk; ++b) s += partial[(size_t)b * 4160 + kk * 64 + col];
  } else if (col == 64) {
    for (int b = 0; b < nblk; ++b) s += partial[(size_t)b * 4160 + 4096 + kk];
  }
  B2[o] = tobf(s);
}

// ---------------------------------------------------------------------------
// Kernel 4: out = (Qphi @ KV) / (Qphi . ksum + eps). 5 N-tiles per 16-row
// tile, 2 k-steps each; tile 4 col 0 carries the denominator.
// ---------------------------------------------------------------------------
__global__ void __launch_bounds__(256) phase2_kernel(
    const __bf16* __restrict__ Qphi,
    const __bf16* __restrict__ B2,
    float* __restrict__ out) {
  const int tid  = threadIdx.x;
  const int lane = tid & 31;
  const int wv   = tid >> 5;
  const int m    = lane & 15;
  const int half = lane >> 4;
  const int tileRow = blockIdx.x * 128 + wv * 16;

  Frag16 Bf[10];
#pragma unroll
  for (int f = 0; f < 10; ++f) {
    const uint4* bp = reinterpret_cast<const uint4*>(B2 + (size_t)(f * 32 + lane) * 16);
    Bf[f].q[0] = bp[0];
    Bf[f].q[1] = bp[1];
  }

  v8f acc[5] = {};
  const __bf16* qrow = Qphi + (size_t)(tileRow + m) * D_K;
#pragma unroll
  for (int ks = 0; ks < 2; ++ks) {
    Frag16 A;
#pragma unroll
    for (int g = 0; g < 2; ++g)
      A.q[g] = *reinterpret_cast<const uint4*>(qrow + ks * 32 + g * 16 + half * 8);
#pragma unroll
    for (int nt = 0; nt < 5; ++nt)
      acc[nt] = __builtin_amdgcn_wmma_f32_16x16x32_bf16(
          false, A.v, false, Bf[nt * 2 + ks].v, (short)0, acc[nt], false, false);
  }

#pragma unroll
  for (int r = 0; r < 8; ++r) {
    float denom = __shfl(acc[4][r], lane & 16, 32);  // col 64 lives in lane 0/16
    float z = 1.0f / (denom + 1e-8f);
    int row = tileRow + r + 8 * half;
#pragma unroll
    for (int t = 0; t < 4; ++t)
      out[(size_t)row * D_K + t * 16 + m] = acc[t][r] * z;
  }
}

// ---------------------------------------------------------------------------
extern "C" void kernel_launch(void* const* d_in, const int* in_sizes, int n_in,
                              void* d_out, int out_size, void* d_ws, size_t ws_size,
                              hipStream_t stream) {
  (void)n_in; (void)out_size; (void)ws_size;
  const float* X  = (const float*)d_in[0];
  const float* Wq = (const float*)d_in[1];
  const float* Wk = (const float*)d_in[2];
  const float* Wv = (const float*)d_in[3];
  float* out = (float*)d_out;

  const int n     = in_sizes[0] / D_IN;   // 131072
  const int nblk1 = n / ROWS_P1;          // 512

  char* ws = (char*)d_ws;
  size_t off = 0;
  __bf16* Bfrag = (__bf16*)(ws + off); off += 12 * 16 * 32 * 16 * 2;  // 192 KB
  __bf16* Qphi  = (__bf16*)(ws + off); off += (size_t)n * D_K * 2;    // 16 MB
  float*  part  = (float*)(ws + off);  off += (size_t)nblk1 * 4160 * 4;
  __bf16* B2    = (__bf16*)(ws + off);                                 // 10 KB

  prep_w_kernel<<<(12 * 16 * 32 * 16) / 256, 256, 0, stream>>>(Wq, Wk, Wv, Bfrag);
  phase1_kernel<<<nblk1, 256, 0, stream>>>(X, Bfrag, Qphi, part);
  reduce_kernel<<<(5 * 2 * 32 * 16 + 255) / 256, 256, 0, stream>>>(part, nblk1, B2);
  phase2_kernel<<<n / 128, 256, 0, stream>>>(Qphi, B2, out);
}
